// GraphVAE_87333864997321
// MI455X (gfx1250) — matrix operand order repaired and checked by reading, not compiled, not run on hardware
//
#include <hip/hip_runtime.h>
#include <hip/hip_bf16.h>
#include <math.h>

#define N_NODES 50000
#define N_EDGES 800000
#define IN_DIM  128
#define HID_DIM 64
#define LAT_DIM 32
#define OUT_DIM 64

typedef __attribute__((ext_vector_type(16))) __bf16 v16bf;
typedef __attribute__((ext_vector_type(8)))  float  v8f;

// ---------------------------------------------------------------------------
// Weight packing: fp32 W[KDIM x F] (row-major) -> bf16 B fragments in this GEMM's
// lane order (k-in-block = 16*(lane>>4) + i, as for A), so each lane loads its 16-element fragment as two
// contiguous global_load_b128.
//
// packed[((kblk*(F/16) + tileN)*32 + lane)*16 + i] =
//      bf16( W[(kblk*32 + (lane>>4)*16 + i) * F + tileN*16 + (lane&15)] )
// ---------------------------------------------------------------------------
template <int KDIM, int F>
__global__ void pack_weight_kernel(const float* __restrict__ W,
                                   __bf16* __restrict__ packed)
{
    int t = blockIdx.x * blockDim.x + threadIdx.x;     // [0, KDIM*F)
    if (t >= KDIM * F) return;
    const int i     = t & 15;
    const int lane  = (t >> 4) & 31;
    const int tileN = (t >> 9) % (F / 16);
    const int kblk  = t / (512 * (F / 16));
    const int k     = kblk * 32 + (lane >> 4) * 16 + i;
    const int col   = tileN * 16 + (lane & 15);
    packed[t] = (__bf16)W[k * F + col];
}

// ---------------------------------------------------------------------------
// WMMA GEMM: C[rows x F] = A[rows x KDIM] @ W[KDIM x F] (+ bias optional)
// One wave computes a full 16-row stripe: the A fragment for each K-step is
// loaded once and reused across all F/16 column tiles (F/16 accumulators).
// rows must be a multiple of 16 (50000 = 3125*16).
// ---------------------------------------------------------------------------
template <int KDIM, int F>
__global__ __launch_bounds__(32)
void gemm_bf16_wmma(const float*  __restrict__ A,
                    const __bf16* __restrict__ Wp,     // packed fragments
                    const float*  __restrict__ bias,   // may be nullptr
                    float* __restrict__ C)
{
    constexpr int NT = F / 16;                // column tiles per stripe
    const int lane = threadIdx.x & 31;
    const int half = lane >> 4;               // 0: lanes 0-15, 1: lanes 16-31
    const int l16  = lane & 15;
    const int row  = blockIdx.x * 16 + l16;   // A-fragment row (M = l16)
    const int kbA  = half * 16;               // A K-group offset for this half

    const float* __restrict__ arow = A + (size_t)row * KDIM;

    v8f acc[NT];
#pragma unroll
    for (int n = 0; n < NT; ++n) acc[n] = (v8f){};

#pragma unroll
    for (int k0 = 0; k0 < KDIM; k0 += 32) {
        const int kblk = k0 / 32;
        // A 16x32 bf16 fragment: slot i -> K = kbA + i
        v16bf afrag;
#pragma unroll
        for (int i = 0; i < 16; ++i) {
            const int ka = kbA + i;
            afrag[i] = (__bf16)arow[k0 + ka];
        }
        // One B fragment per column tile, A reused from registers
#pragma unroll
        for (int n = 0; n < NT; ++n) {
            const v16bf bfrag = *(const v16bf*)(Wp +
                (((size_t)(kblk * NT + n) * 32 + lane) * 16));
            acc[n] = __builtin_amdgcn_wmma_f32_16x16x32_bf16(
                false, afrag, false, bfrag, (short)0, acc[n], false, false);
        }
    }

    // C/D 16x16 f32: VGPR r -> M = r + 8*half, N = l16
#pragma unroll
    for (int n = 0; n < NT; ++n) {
        const int col = n * 16 + l16;
        const float bv = bias ? bias[col] : 0.0f;
#pragma unroll
        for (int r = 0; r < 8; ++r) {
            const int orow = blockIdx.x * 16 + half * 8 + r;
            C[(size_t)orow * F + col] = acc[n][r] + bv;
        }
    }
}

// ---------------------------------------------------------------------------
// Degree (with self loop): deg[n] = 1 + #incoming edges
// ---------------------------------------------------------------------------
__global__ void deg_init_kernel(float* __restrict__ deg)
{
    int i = blockIdx.x * blockDim.x + threadIdx.x;
    if (i < N_NODES) deg[i] = 1.0f;
}

__global__ void deg_count_kernel(const int* __restrict__ dst, float* __restrict__ deg)
{
    int e = blockIdx.x * blockDim.x + threadIdx.x;
    if (e < N_EDGES) atomicAdd(&deg[dst[e]], 1.0f);
}

// ---------------------------------------------------------------------------
// Aggregation, float4-vectorized (F is 32 or 64 -> shifts, b128 accesses):
//   acc = hlin / deg            (self-loop term)
//   acc[dst] += hlin[src]*norm  (edge scatter; b128 gather + 4 f32 atomics)
//   out = act(acc + bias)
// ---------------------------------------------------------------------------
template <int F>
__global__ void agg_self_kernel(const float* __restrict__ hlin,
                                const float* __restrict__ deg,
                                float* __restrict__ acc)
{
    constexpr int Q = F / 4;
    size_t t = (size_t)blockIdx.x * blockDim.x + threadIdx.x;
    if (t >= (size_t)N_NODES * Q) return;
    const int   node = (int)(t / Q);
    const float rd   = 1.0f / deg[node];
    float4 h = *(const float4*)(hlin + t * 4);
    h.x *= rd; h.y *= rd; h.z *= rd; h.w *= rd;
    *(float4*)(acc + t * 4) = h;
}

template <int F>
__global__ void agg_edges_kernel(const int* __restrict__ src,
                                 const int* __restrict__ dst,
                                 const float* __restrict__ hlin,
                                 const float* __restrict__ deg,
                                 float* __restrict__ acc)
{
    constexpr int Q = F / 4;                  // quads per edge
    size_t t = (size_t)blockIdx.x * blockDim.x + threadIdx.x;
    if (t >= (size_t)N_EDGES * Q) return;
    const int e = (int)(t / Q);
    const int q = (int)(t % Q);
    const int s = src[e];
    const int d = dst[e];
    const float norm = rsqrtf(deg[s]) * rsqrtf(deg[d]);
    const float4 h = *(const float4*)(hlin + (size_t)s * F + q * 4);
    float* a = acc + (size_t)d * F + q * 4;
    atomicAdd(a + 0, h.x * norm);
    atomicAdd(a + 1, h.y * norm);
    atomicAdd(a + 2, h.z * norm);
    atomicAdd(a + 3, h.w * norm);
}

// ACT: 0 = none, 1 = relu, 2 = sigmoid
template <int F, int ACT>
__global__ void agg_finish_kernel(const float* __restrict__ acc,
                                  const float* __restrict__ bias,
                                  float* __restrict__ out)
{
    constexpr int Q = F / 4;
    size_t t = (size_t)blockIdx.x * blockDim.x + threadIdx.x;
    if (t >= (size_t)N_NODES * Q) return;
    const int q = (int)(t % Q);
    const float4 bv = *(const float4*)(bias + q * 4);
    float4 v = *(const float4*)(acc + t * 4);
    v.x += bv.x; v.y += bv.y; v.z += bv.z; v.w += bv.w;
    if (ACT == 1) {
        v.x = fmaxf(v.x, 0.0f); v.y = fmaxf(v.y, 0.0f);
        v.z = fmaxf(v.z, 0.0f); v.w = fmaxf(v.w, 0.0f);
    } else if (ACT == 2) {
        v.x = 1.0f / (1.0f + __expf(-v.x));
        v.y = 1.0f / (1.0f + __expf(-v.y));
        v.z = 1.0f / (1.0f + __expf(-v.z));
        v.w = 1.0f / (1.0f + __expf(-v.w));
    }
    *(float4*)(out + t * 4) = v;
}

// ---------------------------------------------------------------------------
// Reparameterize: z = mu + eps * exp(0.5 * log_var), float4-vectorized
// ---------------------------------------------------------------------------
__global__ void reparam_kernel(const float* __restrict__ mu,
                               const float* __restrict__ lv,
                               const float* __restrict__ eps,
                               float* __restrict__ z)
{
    size_t t = (size_t)blockIdx.x * blockDim.x + threadIdx.x;
    if (t >= (size_t)N_NODES * LAT_DIM / 4) return;
    const float4 m = *(const float4*)(mu  + t * 4);
    const float4 l = *(const float4*)(lv  + t * 4);
    const float4 e = *(const float4*)(eps + t * 4);
    float4 r;
    r.x = m.x + e.x * __expf(0.5f * l.x);
    r.y = m.y + e.y * __expf(0.5f * l.y);
    r.z = m.z + e.z * __expf(0.5f * l.z);
    r.w = m.w + e.w * __expf(0.5f * l.w);
    *(float4*)(z + t * 4) = r;
}

// ---------------------------------------------------------------------------
// Host-side helpers
// ---------------------------------------------------------------------------
template <int F, int ACT>
static inline void run_agg(const int* src, const int* dst,
                           const float* hlin, const float* deg, float* acc,
                           const float* bias, float* outbuf, hipStream_t stream)
{
    size_t nq = (size_t)N_NODES * (F / 4);
    size_t eq = (size_t)N_EDGES * (F / 4);
    agg_self_kernel<F><<<(unsigned)((nq + 255) / 256), 256, 0, stream>>>(hlin, deg, acc);
    agg_edges_kernel<F><<<(unsigned)((eq + 255) / 256), 256, 0, stream>>>(src, dst, hlin, deg, acc);
    agg_finish_kernel<F, ACT><<<(unsigned)((nq + 255) / 256), 256, 0, stream>>>(acc, bias, outbuf);
}

template <int KDIM, int F>
static inline void run_gemm(const float* A, const float* W, __bf16* wpack,
                            const float* bias, float* C, hipStream_t stream)
{
    pack_weight_kernel<KDIM, F><<<(KDIM * F + 255) / 256, 256, 0, stream>>>(W, wpack);
    gemm_bf16_wmma<KDIM, F><<<N_NODES / 16, 32, 0, stream>>>(A, wpack, bias, C);
}

extern "C" void kernel_launch(void* const* d_in, const int* in_sizes, int n_in,
                              void* d_out, int out_size, void* d_ws, size_t ws_size,
                              hipStream_t stream)
{
    // Inputs in setup_inputs() order
    const float* x    = (const float*)d_in[0];
    const int*   ei   = (const int*)  d_in[1];   // edge_index [2, E]
    const float* eps  = (const float*)d_in[2];
    const float* W1   = (const float*)d_in[3];
    const float* b1   = (const float*)d_in[4];
    const float* W2   = (const float*)d_in[5];
    const float* b2   = (const float*)d_in[6];
    const float* Wmu  = (const float*)d_in[7];
    const float* bmu  = (const float*)d_in[8];
    const float* Wlv  = (const float*)d_in[9];
    const float* blv  = (const float*)d_in[10];
    const float* Wd1  = (const float*)d_in[11];
    const float* bd1  = (const float*)d_in[12];
    const float* Wd2  = (const float*)d_in[13];
    const float* bd2  = (const float*)d_in[14];
    (void)in_sizes; (void)n_in; (void)out_size; (void)ws_size;

    const int* src = ei;              // row 0
    const int* dst = ei + N_EDGES;    // row 1

    // Outputs: recon [N,64] | mu [N,32] | log_var [N,32]
    float* out   = (float*)d_out;
    float* recon = out;
    float* mu    = out + (size_t)N_NODES * OUT_DIM;
    float* lv    = mu  + (size_t)N_NODES * LAT_DIM;

    // Workspace carve-up (floats): deg | hlin(64) | acc(64) | featA(64) | featB(32) | wpack
    // All offsets are multiples of 16 bytes -> float4 accesses stay aligned.
    float* ws    = (float*)d_ws;
    float* deg   = ws;  ws += N_NODES;
    float* hlin  = ws;  ws += (size_t)N_NODES * 64;
    float* acc   = ws;  ws += (size_t)N_NODES * 64;
    float* featA = ws;  ws += (size_t)N_NODES * 64;
    float* featB = ws;  ws += (size_t)N_NODES * 32;
    __bf16* wpack = (__bf16*)ws;      // max 128*64 bf16 = 16 KB, reused serially

    // ---- degree (with self loops) ----
    deg_init_kernel<<<(N_NODES + 255) / 256, 256, 0, stream>>>(deg);
    deg_count_kernel<<<(N_EDGES + 255) / 256, 256, 0, stream>>>(dst, deg);

    // ---- encoder layer 1: h1 = relu(gcn(x, W1, b1)), 128 -> 64 ----
    run_gemm<IN_DIM, HID_DIM>(x, W1, wpack, nullptr, hlin, stream);
    run_agg<HID_DIM, 1>(src, dst, hlin, deg, acc, b1, featA, stream);

    // ---- encoder layer 2: h2 = relu(gcn(h1, W2, b2)), 64 -> 32 ----
    run_gemm<HID_DIM, LAT_DIM>(featA, W2, wpack, nullptr, hlin, stream);
    run_agg<LAT_DIM, 1>(src, dst, hlin, deg, acc, b2, featB, stream);

    // ---- mu / log_var heads (plain linear + bias), 32 -> 32 ----
    run_gemm<LAT_DIM, LAT_DIM>(featB, Wmu, wpack, bmu, mu, stream);
    run_gemm<LAT_DIM, LAT_DIM>(featB, Wlv, wpack, blv, lv, stream);

    // ---- reparameterize: z = mu + eps * exp(0.5*lv) (reuse featB) ----
    {
        size_t nz = (size_t)N_NODES * LAT_DIM / 4;
        reparam_kernel<<<(unsigned)((nz + 255) / 256), 256, 0, stream>>>(mu, lv, eps, featB);
    }

    // ---- decoder layer 1: d = relu(gcn(z, Wd1, bd1)), 32 -> 64 ----
    run_gemm<LAT_DIM, HID_DIM>(featB, Wd1, wpack, nullptr, hlin, stream);
    run_agg<HID_DIM, 1>(src, dst, hlin, deg, acc, bd1, featA, stream);

    // ---- decoder layer 2: recon = sigmoid(gcn(d, Wd2, bd2)), 64 -> 64 ----
    run_gemm<HID_DIM, OUT_DIM>(featA, Wd2, wpack, nullptr, hlin, stream);
    run_agg<OUT_DIM, 2>(src, dst, hlin, deg, acc, bd2, recon, stream);
}
